// ModuleBatchedExperts_15659450761318
// MI455X (gfx1250) — compile-verified, matrix operand order checked by reading, not run on hardware
//
#include <hip/hip_runtime.h>
#include <hip/hip_bf16.h>

// ---------------------------------------------------------------------------
// MoE top-2 batched experts for MI455X (gfx1250, wave32, WMMA).
//   out[n] = sum_e routing[n,e] * ( gelu(x[n] @ W1[e] + b1[e]) @ W2[e] + b2[e] )
// Sparse path: gather tokens per expert; bf16 WMMA 16x16x32 (f32 accum).
// Transposed formulation (weights = A operand, tokens = B operand): GEMM1's
// C-fragment packs into one ds_store_b128 (8 consecutive hidden values per
// lane) and the final epilogue owns one token per lane. 16 waves per block
// keep peak VGPR use < 256 (no s_set_vgpr_msb in the hot loops).
// ---------------------------------------------------------------------------

#define NDIM   512
#define NHID   1024
#define NEXP   16
#define NTOK   8192
#define TILE_M 64                 // tokens per block tile (4 WMMA N-tiles)
#define HCHUNK 512                // hidden chunk resident in LDS
#define NTILES (NTOK / TILE_M)    // 128 tiles per expert (worst-case coverage)
#define BLK    512                // 16 waves

typedef __attribute__((ext_vector_type(16))) __bf16 v16bf;
typedef __attribute__((ext_vector_type(8)))  __bf16 bf8;
typedef __attribute__((ext_vector_type(4)))  __bf16 bf4;
typedef __attribute__((ext_vector_type(8)))  float  v8f;

union FragU { v16bf v; bf8 h[2]; };

// One WMMA fragment from a row-major matrix with K contiguous.
// ISA layout: lane&15 = row(A)/col(B); lanes>=16 hold K+8; vector elements
// 8..15 hold K+16..23. Caller pre-applies row*ld + ((lane>>4)<<3).
__device__ __forceinline__ v16bf ldfrag(const __bf16* q) {
  FragU f;
  f.h[0] = *(const bf8*)(q);
  f.h[1] = *(const bf8*)(q + 16);
  return f.v;
}

__device__ __forceinline__ float gelu_exact(float v) {
  return 0.5f * v * (1.0f + erff(v * 0.70710678118654752f));
}

#define WMMA_BF16(A, B, C) \
  __builtin_amdgcn_wmma_f32_16x16x32_bf16(false, (A), false, (B), (short)0, (C), false, false)

// ---------------------------------------------------------------------------
// K0: zero output (atomics accumulate into it) and per-expert counters.
__global__ void moe_zero(float* __restrict__ out, int* __restrict__ counts) {
  int gid = blockIdx.x * 256 + threadIdx.x;       // < NTOK*NDIM/4
  float4 z; z.x = 0.f; z.y = 0.f; z.z = 0.f; z.w = 0.f;
  ((float4*)out)[gid] = z;
  if (gid < NEXP) counts[gid] = 0;
}

// K1: fp32 weights -> bf16, transposed so A-operand rows have K contiguous.
//   W1[e][d][h] -> W1t[e][h][d]   (GEMM1 A row = hidden index, K = d)
//   W2[e][h][d] -> W2t[e][d][h]   (GEMM2 A row = output dim,   K = h)
__global__ void moe_convert_weights(const float* __restrict__ W1,
                                    const float* __restrict__ W2,
                                    __bf16* __restrict__ W1t,
                                    __bf16* __restrict__ W2t) {
  int gid = blockIdx.x * 256 + threadIdx.x;       // < NEXP*NDIM*NHID
  int e   = gid / (NDIM * NHID);
  int rem = gid - e * (NDIM * NHID);
  int d1  = rem / NHID, h1 = rem % NHID;
  W1t[((size_t)e * NHID + h1) * NDIM + d1] = (__bf16)W1[gid];
  int h2  = rem / NDIM, d2 = rem % NDIM;
  W2t[((size_t)e * NDIM + d2) * NHID + h2] = (__bf16)W2[gid];
}

// K2: compact routing into per-expert (token id, weight) lists.
__global__ void moe_route(const float* __restrict__ routing,
                          int* __restrict__ counts,
                          int* __restrict__ tok_id,
                          float* __restrict__ tok_w) {
  int gid = blockIdx.x * 256 + threadIdx.x;       // < NTOK*NEXP
  float v = routing[gid];
  if (v > 0.0f) {
    int n = gid >> 4, e = gid & 15;
    int pos = atomicAdd(counts + e, 1);
    tok_id[e * NTOK + pos] = n;
    tok_w [e * NTOK + pos] = v;
  }
}

// K3: fused expert MLP for one (expert, 64-token tile).
// 512 threads = 16 waves. LDS: As[64][512]bf16 (64KB) + Hs[64][512]bf16 (64KB).
// Computed transposed: Hs^T = gelu(W1t . As^T), Y^T = W2t . Hs^T.
// Wave owns hidden tiles {wid, wid+16} (GEMM1) and dim tiles {wid, wid+16}
// (GEMM2): 8 persistent accumulators -> peak VGPRs stay below 256.
__global__ __launch_bounds__(BLK)
void moe_expert(const float* __restrict__ x,
                const float* __restrict__ b1,
                const float* __restrict__ b2,
                const int*   __restrict__ counts,
                const int*   __restrict__ tok_id,
                const float* __restrict__ tok_w,
                const __bf16* __restrict__ W1t,
                const __bf16* __restrict__ W2t,
                float* __restrict__ out) {
  const int e  = blockIdx.x / NTILES;
  const int t  = blockIdx.x % NTILES;
  const int m0 = t * TILE_M;
  const int cnt = counts[e];
  if (m0 >= cnt) return;                      // uniform per block: safe early exit
  const int rows = min(TILE_M, cnt - m0);

  extern __shared__ __bf16 smem[];
  __bf16* As = smem;                          // [TILE_M][NDIM]   tokens (B operand)
  __bf16* Hs = smem + TILE_M * NDIM;          // [TILE_M][HCHUNK] hidden (B operand)

  const int tid  = threadIdx.x;
  const int lane = tid & 31;
  const int wid  = tid >> 5;                  // 0..15
  const int ln   = lane & 15;                 // fragment row/col
  const int khi  = (lane >> 4) << 3;          // K offset for upper half-wave
  const int* tids = tok_id + e * NTOK + m0;
  const float* tws = tok_w + e * NTOK + m0;

  // ---- stage gathered x rows -> bf16 LDS (pad short tiles with zeros) ----
  #pragma unroll
  for (int j = 0; j < (TILE_M * NDIM / 4) / BLK; ++j) {
    int idx = tid + j * BLK;                  // float4 index
    int row = idx >> 7;                       // NDIM/4 = 128 float4 per row
    int c4  = idx & 127;
    float4 v; v.x = 0.f; v.y = 0.f; v.z = 0.f; v.w = 0.f;
    if (row < rows) {
      int tok = tids[row];
      v = ((const float4*)(x + (size_t)tok * NDIM))[c4];
    }
    bf4 b; b[0] = (__bf16)v.x; b[1] = (__bf16)v.y; b[2] = (__bf16)v.z; b[3] = (__bf16)v.w;
    *(bf4*)(As + (size_t)row * NDIM + c4 * 4) = b;
  }
  __syncthreads();

  // persistent GEMM2 accumulators: [dim tile p(2)][token tile nt(4)] -> 8 x v8f
  v8f acc2[8];
  #pragma unroll
  for (int i = 0; i < 8; ++i)
    #pragma unroll
    for (int r = 0; r < 8; ++r) acc2[i][r] = 0.0f;

  const __bf16* W1e = W1t + (size_t)e * NHID * NDIM;
  const __bf16* W2e = W2t + (size_t)e * NDIM * NHID;

  #pragma unroll 1
  for (int hc = 0; hc < NHID / HCHUNK; ++hc) {
    if (hc) __syncthreads();                  // Hs fully consumed before rewrite

    // ---- GEMM1: Hs^T = gelu(W1t[chunk rows] . As^T + b1) ------------------
    // Wave owns hidden tiles {wid, wid+16}. Each weight frag feeds 4 WMMAs.
    {
      v8f cc[2][4];
      #pragma unroll
      for (int p = 0; p < 2; ++p)
        #pragma unroll
        for (int nt = 0; nt < 4; ++nt)
          #pragma unroll
          for (int r = 0; r < 8; ++r) cc[p][nt][r] = 0.0f;

      const __bf16* a0p = W1e + (size_t)(hc * HCHUNK + wid * 16 + ln) * NDIM + khi;
      const __bf16* a1p = a0p + (size_t)(16 * 16) * NDIM;   // +16 tiles of rows
      const __bf16* bp  = As + (size_t)ln * NDIM + khi;
      #pragma unroll 2
      for (int k = 0; k < NDIM; k += 32) {
        v16bf a0 = ldfrag(a0p + k);
        v16bf a1 = ldfrag(a1p + k);
        v16bf t0 = ldfrag(bp + k);
        v16bf t1 = ldfrag(bp + 16 * NDIM + k);
        v16bf t2 = ldfrag(bp + 32 * NDIM + k);
        v16bf t3 = ldfrag(bp + 48 * NDIM + k);
        cc[0][0] = WMMA_BF16(a0, t0, cc[0][0]);
        cc[0][1] = WMMA_BF16(a0, t1, cc[0][1]);
        cc[0][2] = WMMA_BF16(a0, t2, cc[0][2]);
        cc[0][3] = WMMA_BF16(a0, t3, cc[0][3]);
        cc[1][0] = WMMA_BF16(a1, t0, cc[1][0]);
        cc[1][1] = WMMA_BF16(a1, t1, cc[1][1]);
        cc[1][2] = WMMA_BF16(a1, t2, cc[1][2]);
        cc[1][3] = WMMA_BF16(a1, t3, cc[1][3]);
      }
      // epilogue: +b1, GELU, packed ds_store_b128 (8 consecutive h per lane)
      #pragma unroll
      for (int p = 0; p < 2; ++p) {
        const int hbase = (wid + 16 * p) * 16 + khi;      // hidden offset in chunk
        const float* bb = b1 + e * NHID + hc * HCHUNK + hbase;
        float4 bv0 = *(const float4*)(bb);
        float4 bv1 = *(const float4*)(bb + 4);
        #pragma unroll
        for (int nt = 0; nt < 4; ++nt) {
          bf8 pk;
          pk[0] = (__bf16)gelu_exact(cc[p][nt][0] + bv0.x);
          pk[1] = (__bf16)gelu_exact(cc[p][nt][1] + bv0.y);
          pk[2] = (__bf16)gelu_exact(cc[p][nt][2] + bv0.z);
          pk[3] = (__bf16)gelu_exact(cc[p][nt][3] + bv0.w);
          pk[4] = (__bf16)gelu_exact(cc[p][nt][4] + bv1.x);
          pk[5] = (__bf16)gelu_exact(cc[p][nt][5] + bv1.y);
          pk[6] = (__bf16)gelu_exact(cc[p][nt][6] + bv1.z);
          pk[7] = (__bf16)gelu_exact(cc[p][nt][7] + bv1.w);
          *(bf8*)(Hs + (size_t)(nt * 16 + ln) * HCHUNK + hbase) = pk;
        }
      }
    }
    __syncthreads();

    // ---- GEMM2 partial: acc2 += W2t[dim tile rows] . Hs^T -----------------
    // 2 weight frags + 4 token frags per k-step -> 8 WMMAs.
    {
      const __bf16* a0p = W2e + (size_t)(wid * 16 + ln) * NHID + hc * HCHUNK + khi;
      const __bf16* a1p = a0p + (size_t)(16 * 16) * NHID;  // +16 tiles of rows
      const __bf16* bp  = Hs + (size_t)ln * HCHUNK + khi;
      #pragma unroll 2
      for (int k = 0; k < HCHUNK; k += 32) {
        v16bf a0 = ldfrag(a0p + k);
        v16bf a1 = ldfrag(a1p + k);
        v16bf t0 = ldfrag(bp + k);
        v16bf t1 = ldfrag(bp + 16 * HCHUNK + k);
        v16bf t2 = ldfrag(bp + 32 * HCHUNK + k);
        v16bf t3 = ldfrag(bp + 48 * HCHUNK + k);
        acc2[0] = WMMA_BF16(a0, t0, acc2[0]);
        acc2[1] = WMMA_BF16(a0, t1, acc2[1]);
        acc2[2] = WMMA_BF16(a0, t2, acc2[2]);
        acc2[3] = WMMA_BF16(a0, t3, acc2[3]);
        acc2[4] = WMMA_BF16(a1, t0, acc2[4]);
        acc2[5] = WMMA_BF16(a1, t1, acc2[5]);
        acc2[6] = WMMA_BF16(a1, t2, acc2[6]);
        acc2[7] = WMMA_BF16(a1, t3, acc2[7]);
      }
    }
  }

  // ---- epilogue: +b2, scale by routing weight, atomic scatter-add --------
  // Lane owns token (nt*16+ln); rows r are 8 consecutive output dims.
  #pragma unroll
  for (int nt = 0; nt < 4; ++nt) {
    const int tloc = nt * 16 + ln;
    if (tloc < rows) {
      const int   tok = tids[tloc];
      const float w   = tws[tloc];
      float* orow = out + (size_t)tok * NDIM;
      #pragma unroll
      for (int p = 0; p < 2; ++p) {
        const int dbase = (wid + 16 * p) * 16 + khi;
        const float* bb = b2 + e * NDIM + dbase;
        float4 bv0 = *(const float4*)(bb);
        float4 bv1 = *(const float4*)(bb + 4);
        const v8f a = acc2[p * 4 + nt];
        unsafeAtomicAdd(orow + dbase + 0, (a[0] + bv0.x) * w);
        unsafeAtomicAdd(orow + dbase + 1, (a[1] + bv0.y) * w);
        unsafeAtomicAdd(orow + dbase + 2, (a[2] + bv0.z) * w);
        unsafeAtomicAdd(orow + dbase + 3, (a[3] + bv0.w) * w);
        unsafeAtomicAdd(orow + dbase + 4, (a[4] + bv1.x) * w);
        unsafeAtomicAdd(orow + dbase + 5, (a[5] + bv1.y) * w);
        unsafeAtomicAdd(orow + dbase + 6, (a[6] + bv1.z) * w);
        unsafeAtomicAdd(orow + dbase + 7, (a[7] + bv1.w) * w);
      }
    }
  }
}

// ---------------------------------------------------------------------------
extern "C" void kernel_launch(void* const* d_in, const int* in_sizes, int n_in,
                              void* d_out, int out_size, void* d_ws, size_t ws_size,
                              hipStream_t stream) {
  const float* x       = (const float*)d_in[0];   // [NTOK, NDIM]
  const float* routing = (const float*)d_in[1];   // [NTOK, NEXP]
  const float* W1      = (const float*)d_in[2];   // [NEXP, NDIM, NHID]
  const float* b1      = (const float*)d_in[3];   // [NEXP, NHID]
  const float* W2      = (const float*)d_in[4];   // [NEXP, NHID, NDIM]
  const float* b2      = (const float*)d_in[5];   // [NEXP, NDIM]
  float* out = (float*)d_out;

  // Workspace layout (~34.6 MB):
  //   [0,256)            : int counts[NEXP]
  //   [256, +512K)       : int  tok_id[NEXP][NTOK]
  //   [.., +512K)        : float tok_w[NEXP][NTOK]
  //   [.., +16.78M)      : bf16 W1t[NEXP][NHID][NDIM]
  //   [.., +16.78M)      : bf16 W2t[NEXP][NDIM][NHID]
  char* ws = (char*)d_ws;
  int*    counts = (int*)ws;
  int*    tok_id = (int*)(ws + 256);
  float*  tok_w  = (float*)(ws + 256 + (size_t)NEXP * NTOK * 4);
  __bf16* W1t    = (__bf16*)(ws + 256 + (size_t)NEXP * NTOK * 8);
  __bf16* W2t    = W1t + (size_t)NEXP * NHID * NDIM;

  moe_zero<<<dim3(NTOK * NDIM / 4 / 256), dim3(256), 0, stream>>>(out, counts);
  moe_convert_weights<<<dim3(NEXP * NDIM * NHID / 256), dim3(256), 0, stream>>>(
      W1, W2, W1t, W2t);
  moe_route<<<dim3(NTOK * NEXP / 256), dim3(256), 0, stream>>>(
      routing, counts, tok_id, tok_w);
  moe_expert<<<dim3(NEXP * NTILES), dim3(BLK),
               (TILE_M * NDIM + TILE_M * HCHUNK) * sizeof(__bf16), stream>>>(
      x, b1, b2, counts, tok_id, tok_w, W1t, W2t, out);
}
